// StudentFeatureGate_3435973837515
// MI455X (gfx1250) — matrix operand order repaired and checked by reading, not compiled, not run on hardware
//
#include <hip/hip_runtime.h>
#include <math.h>

typedef __attribute__((ext_vector_type(16))) __bf16 v16bf;
typedef __attribute__((ext_vector_type(8)))  float  v8f;
typedef __attribute__((address_space(3)))    void   lds_void_t;

#define B_ROWS  16384
#define IN_DIM  2048
#define HID_DIM 512
#define E_NUM   64
#define K_TOP   8
#define ALPHA_C 0.01f
#define NOISE_STD_C 1.0f
#define LN_EPS  1e-5f

// CDNA5 async global->LDS copy (ASYNCcnt-tracked), 16 bytes per lane.
__device__ __forceinline__ void async_copy_b128(void* lds, const void* gptr)
{
    asm volatile("global_load_async_to_lds_b128 %0, %1, off"
                 :: "v"((lds_void_t*)lds), "v"(gptr)
                 : "memory");
}
__device__ __forceinline__ void wait_async_le6()
{
    asm volatile("s_wait_asynccnt 0x6" ::: "memory");
}
__device__ __forceinline__ void wait_async_0()
{
    asm volatile("s_wait_asynccnt 0x0" ::: "memory");
}

// WMMA bf16 fragment loads from LDS (row-major tiles, row stride in elements).
// A-frag: lane holds M=lane%16; K chunks {lh*8..+7, 16+lh*8..+7} (lh = lane>>4).
__device__ __forceinline__ v16bf ld_frag_a(const __bf16* row, int kbase, int lh)
{
    union { v16bf v; unsigned u[8]; } f;
    #pragma unroll
    for (int i = 0; i < 8; ++i) {
        int k = kbase + ((i >= 4) ? 16 : 0) + lh * 8 + (i & 3) * 2;
        f.u[i] = *(const unsigned*)(row + k);
    }
    return f.v;
}
// B-frag (from [n][k] tile): lane holds N=lane%16; K chunks {lh*8..+7, 16+lh*8..+7} (same as A).
__device__ __forceinline__ v16bf ld_frag_b(const __bf16* row, int kbase, int lh)
{
    union { v16bf v; unsigned u[8]; } f;
    #pragma unroll
    for (int i = 0; i < 8; ++i)
        f.u[i] = *(const unsigned*)(row + kbase + ((i >= 4) ? 16 : 0) + lh * 8 + (i & 3) * 2);
    return f.v;
}

// ---------------- LayerNorm + cast to bf16 ----------------
__global__ __launch_bounds__(256) void ln_cast_kernel(
    const float* __restrict__ hidden, const float* __restrict__ g,
    const float* __restrict__ b, __bf16* __restrict__ xout)
{
    int row = blockIdx.x;
    int t   = threadIdx.x;
    const float* hr = hidden + (size_t)row * IN_DIM;
    float4 v0 = ((const float4*)hr)[t * 2 + 0];
    float4 v1 = ((const float4*)hr)[t * 2 + 1];
    float s  = v0.x + v0.y + v0.z + v0.w + v1.x + v1.y + v1.z + v1.w;
    float ss = v0.x*v0.x + v0.y*v0.y + v0.z*v0.z + v0.w*v0.w
             + v1.x*v1.x + v1.y*v1.y + v1.z*v1.z + v1.w*v1.w;
    #pragma unroll
    for (int off = 16; off; off >>= 1) {
        s  += __shfl_xor(s,  off, 32);
        ss += __shfl_xor(ss, off, 32);
    }
    __shared__ float red_s[8], red_q[8];
    int wave = t >> 5, lane = t & 31;
    if (lane == 0) { red_s[wave] = s; red_q[wave] = ss; }
    __syncthreads();
    s = 0.f; ss = 0.f;
    #pragma unroll
    for (int i = 0; i < 8; ++i) { s += red_s[i]; ss += red_q[i]; }
    float mean = s * (1.0f / IN_DIM);
    float var  = ss * (1.0f / IN_DIM) - mean * mean;
    float rstd = rsqrtf(var + LN_EPS);

    union { __bf16 h[8]; uint4 u; } pack;
    float xv[8] = {v0.x, v0.y, v0.z, v0.w, v1.x, v1.y, v1.z, v1.w};
    int base = t * 8;
    #pragma unroll
    for (int i = 0; i < 8; ++i) {
        float y = (xv[i] - mean) * rstd * g[base + i] + b[base + i];
        pack.h[i] = (__bf16)y;
    }
    *(uint4*)(xout + (size_t)row * IN_DIM + base) = pack.u;
}

// ------------- weight conversion + transpose: w1t[n][k], w2t[n][k] -------------
__global__ __launch_bounds__(256) void cvt_kernel(
    const float* __restrict__ w1, const float* __restrict__ w2,
    __bf16* __restrict__ w1t, __bf16* __restrict__ w2t)
{
    int i = blockIdx.x * 256 + threadIdx.x;
    if (i < IN_DIM * HID_DIM) {
        int k = i / HID_DIM, n = i % HID_DIM;
        w1t[(size_t)n * IN_DIM + k] = (__bf16)w1[i];
    }
    if (i < HID_DIM * E_NUM) {
        int k = i / E_NUM, n = i % E_NUM;
        w2t[(size_t)n * HID_DIM + k] = (__bf16)w2[i];
    }
}

__global__ void zero_kernel(float* __restrict__ psum, float* __restrict__ cnt)
{
    int i = threadIdx.x;
    if (i < E_NUM) { psum[i] = 0.f; cnt[i] = 0.f; }
}

// ------- GEMM1: x[B,2048] @ w1[2048,512] + b1, GELU -> h bf16 -------
// Block: 256 thr = 8 waves (2M x 4N), tile M=64,N=128, K-chunk 64, async dbl-buffer.
__global__ __launch_bounds__(256) void gemm1_kernel(
    const __bf16* __restrict__ X, const __bf16* __restrict__ W1T,
    const float* __restrict__ b1, __bf16* __restrict__ H)
{
    __shared__ __align__(16) __bf16 sA[2][64][64];    // [buf][m][k]  16KB
    __shared__ __align__(16) __bf16 sB[2][128][64];   // [buf][n][k]  32KB
    int t = threadIdx.x, wave = t >> 5, lane = t & 31;
    int lrow = lane & 15, lh = lane >> 4;
    int n_blk = blockIdx.x * 128;    // fast dim -> adjacent blocks share A rows (L2)
    int m_blk = blockIdx.y * 64;
    int wm = wave & 1;               // 0..1  (M sub-block of 32)
    int wn = wave >> 1;              // 0..3  (N sub-block of 32)
    v8f acc00 = {}, acc01 = {}, acc10 = {}, acc11 = {};

    // copy mapping: 16B chunks, chunk c -> row c>>3, col (c&7)*8
    int cr = t >> 3;                 // 0..31
    int ck = (t & 7) * 8;            // 0..56
    const __bf16* gA0 = X + (size_t)(m_blk + cr) * IN_DIM + ck;
    const __bf16* gB0 = W1T + (size_t)(n_blk + cr) * IN_DIM + ck;

    const int NSTEP = IN_DIM / 64;   // 32
    auto issue = [&](int buf, int step) {
        size_t o = (size_t)step * 64;
        async_copy_b128(&sA[buf][cr][ck],      gA0 + o);
        async_copy_b128(&sA[buf][cr + 32][ck], gA0 + (size_t)32 * IN_DIM + o);
        async_copy_b128(&sB[buf][cr][ck],      gB0 + o);
        async_copy_b128(&sB[buf][cr + 32][ck], gB0 + (size_t)32 * IN_DIM + o);
        async_copy_b128(&sB[buf][cr + 64][ck], gB0 + (size_t)64 * IN_DIM + o);
        async_copy_b128(&sB[buf][cr + 96][ck], gB0 + (size_t)96 * IN_DIM + o);
    };
    issue(0, 0);

    for (int kk = 0; kk < NSTEP; ++kk) {
        int cur = kk & 1;
        if (kk + 1 < NSTEP) {
            issue(1 - cur, kk + 1);
            wait_async_le6();        // this step's 6 copies done; next step's in flight
        } else {
            wait_async_0();
        }
        __syncthreads();

        const __bf16* a0row = &sA[cur][wm * 32 + lrow][0];
        const __bf16* a1row = a0row + 16 * 64;
        const __bf16* b0row = &sB[cur][wn * 32 + lrow][0];
        const __bf16* b1row = b0row + 16 * 64;
        #pragma unroll
        for (int kh = 0; kh < 2; ++kh) {
            int kb = kh * 32;
            v16bf fa0 = ld_frag_a(a0row, kb, lh);
            v16bf fa1 = ld_frag_a(a1row, kb, lh);
            v16bf fb0 = ld_frag_b(b0row, kb, lh);
            v16bf fb1 = ld_frag_b(b1row, kb, lh);
            acc00 = __builtin_amdgcn_wmma_f32_16x16x32_bf16(false, fa0, false, fb0,
                                                            (short)0, acc00, false, false);
            acc01 = __builtin_amdgcn_wmma_f32_16x16x32_bf16(false, fa0, false, fb1,
                                                            (short)0, acc01, false, false);
            acc10 = __builtin_amdgcn_wmma_f32_16x16x32_bf16(false, fa1, false, fb0,
                                                            (short)0, acc10, false, false);
            acc11 = __builtin_amdgcn_wmma_f32_16x16x32_bf16(false, fa1, false, fb1,
                                                            (short)0, acc11, false, false);
        }
        __syncthreads();             // done reading buf[cur] before its next overwrite
    }

    // epilogue: + b1, exact GELU, store bf16
    auto store_tile = [&](v8f acc, int mi, int ni) {
        #pragma unroll
        for (int r = 0; r < 8; ++r) {
            int gm = m_blk + wm * 32 + mi * 16 + lh * 8 + r;
            int gn = n_blk + wn * 32 + ni * 16 + lrow;
            float x = acc[r] + b1[gn];
            float gg = 0.5f * x * (1.0f + erff(x * 0.70710678118f));
            H[(size_t)gm * HID_DIM + gn] = (__bf16)gg;
        }
    };
    store_tile(acc00, 0, 0);
    store_tile(acc01, 0, 1);
    store_tile(acc10, 1, 0);
    store_tile(acc11, 1, 1);
}

// ------- GEMM2: h[B,512] @ w2[512,64] + b2 + noise -> logits f32 -------
__global__ __launch_bounds__(128) void gemm2_kernel(
    const __bf16* __restrict__ H, const __bf16* __restrict__ W2T,
    const float* __restrict__ b2, const float* __restrict__ noise,
    float* __restrict__ logits)
{
    __shared__ __align__(16) __bf16 sA[16][32];
    __shared__ __align__(16) __bf16 sB[64][32];   // [n][k]
    int t = threadIdx.x, wave = t >> 5, lane = t & 31;
    int m_blk = blockIdx.x * 16;
    v8f acc = {};
    int ar = t >> 3, ac = (t & 7) * 4;    // A: 16x32, 4 bf16 (8B)
    int bn2 = t >> 1, bk2 = (t & 1) * 16; // B: 64x32, 16 bf16 (32B)

    for (int kk = 0; kk < HID_DIM; kk += 32) {
        uint2 av = *(const uint2*)(H + (size_t)(m_blk + ar) * HID_DIM + kk + ac);
        *(uint2*)&sA[ar][ac] = av;
        const __bf16* wp = W2T + (size_t)bn2 * HID_DIM + kk + bk2;
        uint4 b0 = *(const uint4*)(wp);
        uint4 b1v = *(const uint4*)(wp + 8);
        *(uint4*)&sB[bn2][bk2]     = b0;
        *(uint4*)&sB[bn2][bk2 + 8] = b1v;
        __syncthreads();

        int lrow = lane & 15, lh = lane >> 4;
        v16bf fa = ld_frag_a(&sA[lrow][0], 0, lh);
        v16bf fb = ld_frag_b(&sB[wave * 16 + lrow][0], 0, lh);
        acc = __builtin_amdgcn_wmma_f32_16x16x32_bf16(false, fa, false, fb,
                                                      (short)0, acc, false, false);
        __syncthreads();
    }
    #pragma unroll
    for (int r = 0; r < 8; ++r) {
        int gm = m_blk + (lane >> 4) * 8 + r;
        int gn = wave * 16 + (lane & 15);
        float v = acc[r] + b2[gn] + NOISE_STD_C * noise[(size_t)gm * E_NUM + gn];
        logits[(size_t)gm * E_NUM + gn] = v;
    }
}

// ------- softmax + top-8 + gating stats (one wave per row) -------
__global__ __launch_bounds__(256) void topk_kernel(
    const float* __restrict__ logits, float* __restrict__ out_idx,
    float* __restrict__ out_w, float* __restrict__ g_psum, float* __restrict__ g_cnt)
{
    __shared__ float l_psum[E_NUM];
    __shared__ float l_cnt[E_NUM];
    int t = threadIdx.x, wave = t >> 5, lane = t & 31;
    if (t < E_NUM) { l_psum[t] = 0.f; l_cnt[t] = 0.f; }
    __syncthreads();

    int row = blockIdx.x * 8 + wave;
    const float* lr = logits + (size_t)row * E_NUM;
    float a0 = lr[lane], a1 = lr[lane + 32];
    float m = fmaxf(a0, a1);
    #pragma unroll
    for (int off = 16; off; off >>= 1) m = fmaxf(m, __shfl_xor(m, off, 32));
    float e0 = __expf(a0 - m), e1 = __expf(a1 - m);
    float s = e0 + e1;
    #pragma unroll
    for (int off = 16; off; off >>= 1) s += __shfl_xor(s, off, 32);
    float inv = 1.0f / s;
    float s0 = e0 * inv, s1 = e1 * inv;
    atomicAdd(&l_psum[lane], s0);
    atomicAdd(&l_psum[lane + 32], s1);

    float c0 = s0, c1 = s1, wsum = 0.f;
    float tv[K_TOP]; int ti[K_TOP];
    #pragma unroll
    for (int j = 0; j < K_TOP; ++j) {
        float v; int idx;
        if (c0 >= c1) { v = c0; idx = lane; } else { v = c1; idx = lane + 32; }
        #pragma unroll
        for (int off = 16; off; off >>= 1) {
            float ov = __shfl_xor(v, off, 32);
            int   oi = __shfl_xor(idx, off, 32);
            if (ov > v || (ov == v && oi < idx)) { v = ov; idx = oi; }
        }
        tv[j] = v; ti[j] = idx; wsum += v;
        if (idx == lane)            c0 = -1.f;
        else if (idx == lane + 32)  c1 = -1.f;
    }
    float wnorm = 1.0f / (wsum + 1e-20f);
    if (lane == 0) {
        #pragma unroll
        for (int j = 0; j < K_TOP; ++j) {
            out_idx[(size_t)row * K_TOP + j] = (float)ti[j];
            out_w  [(size_t)row * K_TOP + j] = tv[j] * wnorm;
            atomicAdd(&l_cnt[ti[j]], 1.f);
        }
    }
    __syncthreads();
    if (t < E_NUM) {
        atomicAdd(&g_psum[t], l_psum[t]);
        atomicAdd(&g_cnt[t],  l_cnt[t]);
    }
}

// ------- aux loss -------
__global__ void aux_kernel(const float* __restrict__ g_psum,
                           const float* __restrict__ g_cnt, float* __restrict__ out)
{
    int t = threadIdx.x;  // 64 threads
    float f = g_cnt[t]  * (1.0f / (float)(B_ROWS * K_TOP));
    float P = g_psum[t] * (1.0f / (float)B_ROWS);
    float v = f * P;
    #pragma unroll
    for (int off = 16; off; off >>= 1) v += __shfl_xor(v, off, 32);
    __shared__ float tmp[2];
    if ((t & 31) == 0) tmp[t >> 5] = v;
    __syncthreads();
    if (t == 0) out[0] = ALPHA_C * (float)E_NUM * (tmp[0] + tmp[1]);
}

extern "C" void kernel_launch(void* const* d_in, const int* in_sizes, int n_in,
                              void* d_out, int out_size, void* d_ws, size_t ws_size,
                              hipStream_t stream) {
    (void)in_sizes; (void)n_in; (void)out_size; (void)ws_size;
    const float* hidden = (const float*)d_in[0];
    const float* ln_g   = (const float*)d_in[1];
    const float* ln_b   = (const float*)d_in[2];
    const float* w1     = (const float*)d_in[3];
    const float* b1     = (const float*)d_in[4];
    const float* w2     = (const float*)d_in[5];
    const float* b2     = (const float*)d_in[6];
    const float* noise  = (const float*)d_in[7];
    float* out = (float*)d_out;

    char* ws = (char*)d_ws;
    size_t off = 0;
    __bf16* xb  = (__bf16*)(ws + off); off += (size_t)B_ROWS * IN_DIM  * 2;
    __bf16* hb  = (__bf16*)(ws + off); off += (size_t)B_ROWS * HID_DIM * 2;
    __bf16* w1t = (__bf16*)(ws + off); off += (size_t)IN_DIM * HID_DIM * 2;
    __bf16* w2t = (__bf16*)(ws + off); off += (size_t)HID_DIM * E_NUM  * 2;
    float* logits = (float*)(ws + off); off += (size_t)B_ROWS * E_NUM * 4;
    float* g_psum = (float*)(ws + off); off += 256;
    float* g_cnt  = (float*)(ws + off); off += 256;

    hipLaunchKernelGGL(zero_kernel, dim3(1), dim3(64), 0, stream, g_psum, g_cnt);
    hipLaunchKernelGGL(cvt_kernel, dim3((IN_DIM * HID_DIM + 255) / 256), dim3(256), 0, stream,
                       w1, w2, w1t, w2t);
    hipLaunchKernelGGL(ln_cast_kernel, dim3(B_ROWS), dim3(256), 0, stream,
                       hidden, ln_g, ln_b, xb);
    hipLaunchKernelGGL(gemm1_kernel, dim3(HID_DIM / 128, B_ROWS / 64), dim3(256), 0, stream,
                       xb, w1t, b1, hb);
    hipLaunchKernelGGL(gemm2_kernel, dim3(B_ROWS / 16), dim3(128), 0, stream,
                       hb, w2t, b2, noise, logits);
    hipLaunchKernelGGL(topk_kernel, dim3(B_ROWS / 8), dim3(256), 0, stream,
                       logits, out, out + (size_t)B_ROWS * K_TOP, g_psum, g_cnt);
    hipLaunchKernelGGL(aux_kernel, dim3(1), dim3(64), 0, stream,
                       g_psum, g_cnt, out + (size_t)2 * B_ROWS * K_TOP);
}
